// PositionAttention_13005160972507
// MI455X (gfx1250) — matrix operand order repaired and checked by reading, not compiled, not run on hardware
//
#include <hip/hip_runtime.h>
#include <hip/hip_bf16.h>

typedef __attribute__((ext_vector_type(16))) __bf16 v16bf;
typedef __attribute__((ext_vector_type(8)))  __bf16 v8bf;
typedef __attribute__((ext_vector_type(8)))  float  v8f;

static constexpr int B_  = 4;
static constexpr int C_  = 512;
static constexpr int N_  = 4096;   // 64*64
static constexpr int CQ_ = 64;

union U16 { v16bf v; v8bf h[2]; };

// ---------------------------------------------------------------------------
// f32 -> bf16 elementwise (weights)
// ---------------------------------------------------------------------------
__global__ void k_cvt_bf16(const float* __restrict__ s, __bf16* __restrict__ d, int n) {
    int i = blockIdx.x * blockDim.x + threadIdx.x;
    int stride = gridDim.x * blockDim.x;
    for (; i < n; i += stride) d[i] = (__bf16)s[i];
}

// ---------------------------------------------------------------------------
// x[B,C,N] f32 -> xt[B,N,C] bf16 (32x32 LDS tile transpose)
// ---------------------------------------------------------------------------
__global__ void k_transpose_cvt(const float* __restrict__ x, __bf16* __restrict__ xt) {
    __shared__ __bf16 tile[32][33];
    int blk = blockIdx.x;
    int nt = blk % (N_ / 32); blk /= (N_ / 32);
    int ct = blk % (C_ / 32); blk /= (C_ / 32);
    int b  = blk;
    int c0 = ct * 32, n0 = nt * 32;
    int t  = threadIdx.x;
    int tn = t & 31, tc = t >> 5;             // tc in 0..7
    const float* xp = x + (((size_t)b * C_ + c0) * N_ + n0);
#pragma unroll
    for (int i = 0; i < 4; i++) {
        int c = tc + i * 8;
        tile[c][tn] = (__bf16)xp[(size_t)c * N_ + tn];
    }
    __syncthreads();
    __bf16* xo = xt + (((size_t)b * N_ + n0) * C_ + c0);
#pragma unroll
    for (int i = 0; i < 4; i++) {
        int n = tc + i * 8;
        xo[(size_t)n * C_ + tn] = tile[tn][n];
    }
}

// ---------------------------------------------------------------------------
// Projection: out[n,o] = sum_c xt[n,c]*w[o,c] + bias[o]
// Each wave: 16 rows x 64 cols. colMajorOut: out[b,o,n] else out[b,n,o].
// ---------------------------------------------------------------------------
__global__ void __launch_bounds__(256)
k_proj(const __bf16* __restrict__ xt, const __bf16* __restrict__ w,
       const float* __restrict__ bias, __bf16* __restrict__ out,
       int Co, int colMajorOut) {
    const int lane = threadIdx.x & 31;
    const int r = lane & 15, hl = lane >> 4;
    int gw = (blockIdx.x * blockDim.x + threadIdx.x) >> 5;
    int ct = Co >> 6;
    int tiles_per_b = (N_ / 16) * ct;
    int b = gw / tiles_per_b;
    int t = gw % tiles_per_b;
    int rowt = t / ct, colt = t % ct;
    int n_base = rowt * 16, o_base = colt * 64;

    v8f acc[4] = {};
    const __bf16* arow = xt + ((size_t)b * N_ + n_base + r) * C_;

    for (int kk = 0; kk < C_; kk += 32) {
        U16 A;
        A.h[0] = *(const v8bf*)(arow + kk + hl * 16);       // K = kk + hl*16 ..
        A.h[1] = *(const v8bf*)(arow + kk + hl * 16 + 8);   // K = kk+hl*16+8 ..
#pragma unroll
        for (int j = 0; j < 4; j++) {
            int o = o_base + j * 16 + r;                    // B: lane = column
            v16bf Bf = *(const v16bf*)(w + (size_t)o * C_ + kk + hl * 16);
            acc[j] = __builtin_amdgcn_wmma_f32_16x16x32_bf16(
                false, A.v, false, Bf, (short)0, acc[j], false, false);
        }
    }
#pragma unroll
    for (int j = 0; j < 4; j++) {
        int o = o_base + j * 16 + r;
        float bv = bias[o];
        if (colMajorOut) {
            v8bf st;
#pragma unroll
            for (int v = 0; v < 8; v++) st[v] = (__bf16)(acc[j][v] + bv);
            *(v8bf*)(out + ((size_t)b * Co + o) * N_ + n_base + hl * 8) = st;
        } else {
#pragma unroll
            for (int v = 0; v < 8; v++) {
                int nn = n_base + hl * 8 + v;
                out[((size_t)b * N_ + nn) * Co + o] = (__bf16)(acc[j][v] + bv);
            }
        }
    }
}

// ---------------------------------------------------------------------------
// Fused flash attention + epilogue. 64-key chunks; row-sum via P @ ones WMMA.
// Block = (b, 128-row n-tile, 128-col c-chunk); wave owns 16 query rows.
// qb[B,N,64], kb[B,N,64] (row per key), vb[B,C,N].
// ---------------------------------------------------------------------------
__global__ void __launch_bounds__(256)
k_attn(const __bf16* __restrict__ qb, const __bf16* __restrict__ kb,
       const __bf16* __restrict__ vb, const float* __restrict__ x,
       const float* __restrict__ gamma, float* __restrict__ out) {
    __shared__ __bf16 ldsP[8][16][72];   // per-wave 16x64 P tile, 144B row stride
    int blk = blockIdx.x;
    int cchunk = blk & 3;  blk >>= 2;
    int ntile  = blk & 31; blk >>= 5;
    int b = blk;
    int wave = threadIdx.x >> 5;
    int lane = threadIdx.x & 31;
    int r = lane & 15, hl = lane >> 4;
    int n_base = ntile * 128 + wave * 16;
    int c_base = cchunk * 128;

    // Persistent Q fragments (16 rows x 64 K, two K=32 steps)
    const __bf16* qrow = qb + ((size_t)b * N_ + n_base + r) * CQ_;
    U16 aQ0, aQ1;
    aQ0.h[0] = *(const v8bf*)(qrow + hl * 16);
    aQ0.h[1] = *(const v8bf*)(qrow + 8 + hl * 16);
    aQ1.h[0] = *(const v8bf*)(qrow + 32 + hl * 16);
    aQ1.h[1] = *(const v8bf*)(qrow + 40 + hl * 16);

    // all-ones B fragment (for matrix-core row sums)
    v16bf onesB;
#pragma unroll
    for (int i = 0; i < 16; i++) onesB[i] = (__bf16)1.0f;

    float m_i[8];
#pragma unroll
    for (int v = 0; v < 8; v++) m_i[v] = -3.0e38f;
    v8f accO[8] = {};
    v8f accL = {};                        // running row sums (softmax denominator)

    const __bf16* kbb = kb + (size_t)b * N_ * CQ_;
    const __bf16* vbb = vb + (size_t)b * C_ * N_;

    for (int m0 = 0; m0 < N_; m0 += 64) {
        // prefetch next key chunk into cache hierarchy
        if (m0 + 64 < N_) {
            __builtin_prefetch(kbb + (size_t)(m0 + 64 + lane) * CQ_, 0, 3);
#pragma unroll
            for (int j = 0; j < 8; j++)
                __builtin_prefetch(vbb + (size_t)(c_base + j * 16 + r) * N_
                                       + m0 + 64 + hl * 32, 0, 3);
        }
        // ---- S = Q @ K^T : four 16x16 tiles (64 keys) ----
        v8f s[4];
#pragma unroll
        for (int mt = 0; mt < 4; mt++) {
            const __bf16* krow = kbb + (size_t)(m0 + mt * 16 + r) * CQ_;
            v16bf b0 = *(const v16bf*)(krow + hl * 16);
            v16bf b1 = *(const v16bf*)(krow + 32 + hl * 16);
            v8f sa = {};
            sa = __builtin_amdgcn_wmma_f32_16x16x32_bf16(false, aQ0.v, false, b0,
                                                         (short)0, sa, false, false);
            sa = __builtin_amdgcn_wmma_f32_16x16x32_bf16(false, aQ1.v, false, b1,
                                                         (short)0, sa, false, false);
            s[mt] = sa;
        }
        // ---- online max (rows in VGPR index, cols across 16 lanes) ----
        float scale[8];
#pragma unroll
        for (int v = 0; v < 8; v++) {
            float mx = fmaxf(fmaxf(s[0][v], s[1][v]), fmaxf(s[2][v], s[3][v]));
#pragma unroll
            for (int mm = 1; mm < 16; mm <<= 1) mx = fmaxf(mx, __shfl_xor(mx, mm, 32));
            float mnew = fmaxf(m_i[v], mx);
            scale[v] = __expf(m_i[v] - mnew);
            m_i[v] = mnew;
#pragma unroll
            for (int mt = 0; mt < 4; mt++) s[mt][v] = __expf(s[mt][v] - mnew);
        }
        // ---- transpose P (D-layout) -> A-layout via per-wave LDS ----
#pragma unroll
        for (int mt = 0; mt < 4; mt++)
#pragma unroll
            for (int v = 0; v < 8; v++)
                ldsP[wave][v + 8 * hl][mt * 16 + r] = (__bf16)s[mt][v];
        U16 aP0, aP1;
        const __bf16* pr = &ldsP[wave][r][0];
        aP0.h[0] = *(const v8bf*)(pr + hl * 16);
        aP0.h[1] = *(const v8bf*)(pr + 8 + hl * 16);
        aP1.h[0] = *(const v8bf*)(pr + 32 + hl * 16);
        aP1.h[1] = *(const v8bf*)(pr + 40 + hl * 16);
        // ---- row sums via matrix core: accL = accL*scale + P @ ones ----
        {
            v8f a = accL;
#pragma unroll
            for (int v = 0; v < 8; v++) a[v] *= scale[v];
            a = __builtin_amdgcn_wmma_f32_16x16x32_bf16(false, aP0.v, false, onesB,
                                                        (short)0, a, false, false);
            accL = __builtin_amdgcn_wmma_f32_16x16x32_bf16(false, aP1.v, false, onesB,
                                                           (short)0, a, false, false);
        }
        // ---- O = O*scale + P @ V (8 column tiles of 16, two K=32 steps) ----
#pragma unroll
        for (int j = 0; j < 8; j++) {
            v8f a = accO[j];
#pragma unroll
            for (int v = 0; v < 8; v++) a[v] *= scale[v];
            const __bf16* vrow = vbb + (size_t)(c_base + j * 16 + r) * N_ + m0 + hl * 16;
            v16bf Bv0 = *(const v16bf*)vrow;
            v16bf Bv1 = *(const v16bf*)(vrow + 32);
            a = __builtin_amdgcn_wmma_f32_16x16x32_bf16(false, aP0.v, false, Bv0,
                                                        (short)0, a, false, false);
            accO[j] = __builtin_amdgcn_wmma_f32_16x16x32_bf16(false, aP1.v, false, Bv1,
                                                              (short)0, a, false, false);
        }
    }
    // ---- epilogue: y = gamma * O/l + x ----
    float g = gamma[0];
#pragma unroll
    for (int j = 0; j < 8; j++) {
        int c = c_base + j * 16 + r;
        size_t base = ((size_t)b * C_ + c) * N_ + n_base + hl * 8;
        v8f xv = *(const v8f*)(x + base);
        v8f o;
#pragma unroll
        for (int v = 0; v < 8; v++) o[v] = g * (accO[j][v] / accL[v]) + xv[v];
        *(v8f*)(out + base) = o;
    }
}

// ---------------------------------------------------------------------------
extern "C" void kernel_launch(void* const* d_in, const int* in_sizes, int n_in,
                              void* d_out, int out_size, void* d_ws, size_t ws_size,
                              hipStream_t stream) {
    const float* x     = (const float*)d_in[0];
    const float* q_w   = (const float*)d_in[1];
    const float* q_b   = (const float*)d_in[2];
    const float* k_w   = (const float*)d_in[3];
    const float* k_b   = (const float*)d_in[4];
    const float* v_w   = (const float*)d_in[5];
    const float* v_b   = (const float*)d_in[6];
    const float* gamma = (const float*)d_in[7];
    float* out = (float*)d_out;

    char* ws = (char*)d_ws;
    __bf16* xt   = (__bf16*)ws; ws += (size_t)B_ * N_ * C_ * 2;   // 16.8 MB
    __bf16* qbuf = (__bf16*)ws; ws += (size_t)B_ * N_ * CQ_ * 2;  //  2.1 MB
    __bf16* kbuf = (__bf16*)ws; ws += (size_t)B_ * N_ * CQ_ * 2;  //  2.1 MB
    __bf16* vbuf = (__bf16*)ws; ws += (size_t)B_ * C_ * N_ * 2;   // 16.8 MB
    __bf16* qwb  = (__bf16*)ws; ws += (size_t)CQ_ * C_ * 2;
    __bf16* kwb  = (__bf16*)ws; ws += (size_t)CQ_ * C_ * 2;
    __bf16* vwb  = (__bf16*)ws; ws += (size_t)C_ * C_ * 2;

    // weights -> bf16
    k_cvt_bf16<<<64, 256, 0, stream>>>(q_w, qwb, CQ_ * C_);
    k_cvt_bf16<<<64, 256, 0, stream>>>(k_w, kwb, CQ_ * C_);
    k_cvt_bf16<<<256, 256, 0, stream>>>(v_w, vwb, C_ * C_);
    // x -> xt (transposed bf16)
    k_transpose_cvt<<<B_ * (C_ / 32) * (N_ / 32), 256, 0, stream>>>(x, xt);
    // projections
    k_proj<<<(B_ * (N_ / 16) * 1) / 8, 256, 0, stream>>>(xt, qwb, q_b, qbuf, CQ_, 0);
    k_proj<<<(B_ * (N_ / 16) * 1) / 8, 256, 0, stream>>>(xt, kwb, k_b, kbuf, CQ_, 0);
    k_proj<<<(B_ * (N_ / 16) * 8) / 8, 256, 0, stream>>>(xt, vwb, v_b, vbuf, C_, 1);
    // fused attention
    k_attn<<<B_ * (N_ / 128) * (C_ / 128), 256, 0, stream>>>(qbuf, kbuf, vbuf, x, gamma, out);
}
